// GATModel_70162585748132
// MI455X (gfx1250) — matrix operand order repaired and checked by reading, not compiled, not run on hardware
//
#include <hip/hip_runtime.h>

typedef __attribute__((ext_vector_type(16))) _Float16 v16h;
typedef __attribute__((ext_vector_type(8)))  _Float16 v8h;
typedef __attribute__((ext_vector_type(8)))  float    v8f;

#define NN 50000
#define NE 800000
#define NH 4

// ---------- ordered-uint encoding for float atomic max ----------
__device__ __forceinline__ unsigned f2ord(float f) {
  unsigned u = __float_as_uint(f);
  return (u & 0x80000000u) ? ~u : (u | 0x80000000u);
}
__device__ __forceinline__ float ord2f(unsigned u) {
  return __uint_as_float((u & 0x80000000u) ? (u & 0x7fffffffu) : ~u);
}

// ---------- weight transpose + f32->f16 convert: Wt[n*K+k] = W[k*N+n] ----------
__global__ void k_transpose_w(const float* __restrict__ W, _Float16* __restrict__ Wt,
                              int K, int N) {
  int i = blockIdx.x * blockDim.x + threadIdx.x;
  if (i >= K * N) return;
  int n = i / K, k = i % K;
  Wt[i] = (_Float16)W[(size_t)k * N + n];
}

// ---------- elementwise f32 -> f16 ----------
__global__ void k_cvt_f16(const float* __restrict__ a, _Float16* __restrict__ o, int n) {
  int i = blockIdx.x * blockDim.x + threadIdx.x;
  if (i < n) o[i] = (_Float16)a[i];
}

// ---------- WMMA GEMM: C[M,N] = Ah[M,K] * Bt^T, Ah f16 row-major, Bt f16 [N,K] --
// One wave computes a 16(M) x 64(N) tile: 4 f32 accumulators, K-loop step 32,
// fully unrolled (compile-time K) with software double-buffering of fragments.
// M % 16 == 0, N % 64 == 0, K % 32 == 0 (guaranteed by problem sizes).
template <int K>
__global__ __launch_bounds__(256) void k_wmma_gemm(
    const _Float16* __restrict__ Ah, const _Float16* __restrict__ Bt,
    float* __restrict__ C, int M, int N) {
  const int tilesN = N >> 6;                        // 64-wide column tiles
  const int wave   = (blockIdx.x * blockDim.x + threadIdx.x) >> 5;
  const int lane   = threadIdx.x & 31;
  if (wave >= (M >> 4) * tilesN) return;            // wave-uniform: EXEC stays all-1s

  const int tm = wave / tilesN, tn = wave % tilesN;
  const int row0 = tm << 4;
  const int col0 = tn << 6;
  const int g = lane >> 4, r = lane & 15;

  // A fragment (16-bit A 16x32): lane (g,r) holds M=r, K = kb + g*16 + {0..15}
  const _Float16* pa = Ah + (size_t)(row0 + r) * K + g * 16;
  // B fragment (16-bit B 32x16): lane (g,r) holds N=col, K = kb + g*16 + {0..15}
  const _Float16* pb = Bt + (size_t)(col0 + r) * K + g * 16;

  auto loadA = [&](int kb) -> v16h {
    v8h lo = *(const v8h*)(pa + kb);
    v8h hi = *(const v8h*)(pa + kb + 8);
    return __builtin_shufflevector(lo, hi, 0, 1, 2, 3, 4, 5, 6, 7,
                                   8, 9, 10, 11, 12, 13, 14, 15);
  };
  auto loadB = [&](int nt, int kb) -> v16h {
    return *(const v16h*)(pb + (size_t)nt * 16 * K + kb);
  };

  v8f acc[4] = { {}, {}, {}, {} };

  v16h a0 = loadA(0);
  v16h b0 = loadB(0, 0), b1 = loadB(1, 0), b2 = loadB(2, 0), b3 = loadB(3, 0);

#pragma unroll
  for (int kb = 0; kb < K; kb += 32) {
    v16h an = {}, bn0 = {}, bn1 = {}, bn2 = {}, bn3 = {};
    if (kb + 32 < K) {                              // compile-time resolved
      an  = loadA(kb + 32);
      bn0 = loadB(0, kb + 32);
      bn1 = loadB(1, kb + 32);
      bn2 = loadB(2, kb + 32);
      bn3 = loadB(3, kb + 32);
    }
    acc[0] = __builtin_amdgcn_wmma_f32_16x16x32_f16(false, a0, false, b0, (short)0, acc[0], false, false);
    acc[1] = __builtin_amdgcn_wmma_f32_16x16x32_f16(false, a0, false, b1, (short)0, acc[1], false, false);
    acc[2] = __builtin_amdgcn_wmma_f32_16x16x32_f16(false, a0, false, b2, (short)0, acc[2], false, false);
    acc[3] = __builtin_amdgcn_wmma_f32_16x16x32_f16(false, a0, false, b3, (short)0, acc[3], false, false);
    a0 = an; b0 = bn0; b1 = bn1; b2 = bn2; b3 = bn3;
  }

  // C/D layout: VGPR v, lane (g,r): M = row0 + v + 8*g, N = col0 + nt*16 + r
#pragma unroll
  for (int nt = 0; nt < 4; ++nt) {
#pragma unroll
    for (int v = 0; v < 8; ++v) {
      C[(size_t)(row0 + v + 8 * g) * N + (col0 + nt * 16 + r)] = acc[nt][v];
    }
  }
}

// ---------- per-(node,head) attention coefficients: el = feat.al, er = feat.ar --
__global__ void k_attn_coef(const float* __restrict__ feat,
                            const float* __restrict__ al, const float* __restrict__ ar,
                            float* __restrict__ el, float* __restrict__ er, int D) {
  int i = blockIdx.x * blockDim.x + threadIdx.x;
  if (i >= NN * NH) return;
  int n = i / NH, h = i % NH;
  const float* f = feat + (size_t)n * NH * D + h * D;
  const float* a = al + h * D;
  const float* b = ar + h * D;
  float sl = 0.f, sr = 0.f;
  for (int d = 0; d < D; ++d) { float v = f[d]; sl += v * a[d]; sr += v * b[d]; }
  el[i] = sl; er[i] = sr;
}

// ---------- init per-layer accumulators (ws is poisoned) ----------
__global__ void k_init(float* __restrict__ agg, float* __restrict__ z,
                       unsigned* __restrict__ mu, int aggN) {
  int i = blockIdx.x * blockDim.x + threadIdx.x;
  if (i < aggN) agg[i] = 0.f;
  if (i < NN * NH) { z[i] = 0.f; mu[i] = 0x007FFFFFu; /* f2ord(-inf) */ }
}

// ---------- edge pass 1: segment max of leaky_relu(el[src]+er[dst]) ----------
__global__ void k_edge_max(const int* __restrict__ src, const int* __restrict__ dst,
                           const float* __restrict__ el, const float* __restrict__ er,
                           unsigned* __restrict__ mu) {
  int i = blockIdx.x * blockDim.x + threadIdx.x;
  if (i >= NE * NH) return;
  int e = i / NH, h = i % NH;
  int d = dst[e];
  float v = el[src[e] * NH + h] + er[d * NH + h];
  v = v >= 0.f ? v : 0.2f * v;
  atomicMax(&mu[d * NH + h], f2ord(v));
}

// ---------- edge pass 2: ee = exp(e - m[dst]); z[dst] += ee ----------
__global__ void k_edge_exp(const int* __restrict__ src, const int* __restrict__ dst,
                           const float* __restrict__ el, const float* __restrict__ er,
                           const unsigned* __restrict__ mu,
                           float* __restrict__ ee, float* __restrict__ z) {
  int i = blockIdx.x * blockDim.x + threadIdx.x;
  if (i >= NE * NH) return;
  int e = i / NH, h = i % NH;
  int d = dst[e];
  float v = el[src[e] * NH + h] + er[d * NH + h];
  v = v >= 0.f ? v : 0.2f * v;
  float ex = expf(v - ord2f(mu[d * NH + h]));
  ee[i] = ex;
  atomicAdd(&z[d * NH + h], ex);
}

// ---------- edge pass 3: agg[dst] += feat[src] * (ee / z[dst]) ----------
// one block per edge, blockDim.x == H*D (coalesced feature reads + atomics)
__global__ void k_edge_agg(const int* __restrict__ src, const int* __restrict__ dst,
                           const float* __restrict__ feat, const float* __restrict__ ee,
                           const float* __restrict__ z, float* __restrict__ agg, int D) {
  int e = blockIdx.x;
  int p = threadIdx.x;                 // feature position in [0, H*D)
  int s = src[e], d = dst[e];
  int h = p / D;
  float alpha = ee[(size_t)e * NH + h] / z[d * NH + h];
  atomicAdd(&agg[(size_t)d * NH * D + p], feat[(size_t)s * NH * D + p] * alpha);
}

// ---------- bias + ELU + f16 convert (layer-1 epilogue -> GEMM2 input) ----------
__global__ void k_bias_elu_cvt(const float* __restrict__ a, const float* __restrict__ b,
                               _Float16* __restrict__ o, int total, int F) {
  int i = blockIdx.x * blockDim.x + threadIdx.x;
  if (i >= total) return;
  float v = a[i] + b[i % F];
  v = v > 0.f ? v : (expf(v) - 1.f);
  o[i] = (_Float16)v;
}

// ---------- final: out[n,o] = mean_h (agg2[n,h,o] + b2[h,o]) ----------
__global__ void k_head_mean(const float* __restrict__ agg, const float* __restrict__ b,
                            float* __restrict__ out, int D) {
  int i = blockIdx.x * blockDim.x + threadIdx.x;
  if (i >= NN * D) return;
  int n = i / D, o = i % D;
  float s = 0.f;
  for (int h = 0; h < NH; ++h) s += agg[(size_t)n * NH * D + h * D + o] + b[h * D + o];
  out[i] = s * (1.f / NH);
}

extern "C" void kernel_launch(void* const* d_in, const int* in_sizes, int n_in,
                              void* d_out, int out_size, void* d_ws, size_t ws_size,
                              hipStream_t stream) {
  (void)in_sizes; (void)n_in; (void)out_size; (void)ws_size;
  const float* x   = (const float*)d_in[0];
  const int*   src = (const int*)d_in[1];
  const int*   dst = (const int*)d_in[2];
  const float* W1  = (const float*)d_in[3];
  const float* al1 = (const float*)d_in[4];
  const float* ar1 = (const float*)d_in[5];
  const float* b1  = (const float*)d_in[6];
  const float* W2  = (const float*)d_in[7];
  const float* al2 = (const float*)d_in[8];
  const float* ar2 = (const float*)d_in[9];
  const float* b2  = (const float*)d_in[10];
  float* out = (float*)d_out;

  const int F0 = 128, F1 = 256, F2 = 128, D1 = 64, D2 = 32;

  // -------- workspace carve-out (~145 MB) --------
  size_t off = 0;
  char* base = (char*)d_ws;
  auto carve = [&](size_t bytes) -> void* {
    void* p = base + off; off += (bytes + 255) & ~(size_t)255; return p;
  };
  float*     feat = (float*)carve((size_t)NN * F1 * 4);    // feat1 [N,256] / feat2 [N,128]
  float*     hbuf = (float*)carve((size_t)NN * F1 * 4);    // agg1, later agg2
  _Float16*  ah   = (_Float16*)carve((size_t)NN * F1 * 2); // xh [N,128] / h1h [N,256]
  float*     ee   = (float*)carve((size_t)NE * NH * 4);
  float*     el   = (float*)carve((size_t)NN * NH * 4);
  float*     er   = (float*)carve((size_t)NN * NH * 4);
  float*     zb   = (float*)carve((size_t)NN * NH * 4);
  unsigned*  mu   = (unsigned*)carve((size_t)NN * NH * 4);
  _Float16*  W1t  = (_Float16*)carve((size_t)F0 * F1 * 2); // [256][128] half
  _Float16*  W2t  = (_Float16*)carve((size_t)F1 * F2 * 2); // [128][256] half

  const int B = 256;
  auto cdiv = [](int a, int b) { return (a + b - 1) / b; };

  // -------- weights: transpose + convert to f16 --------
  k_transpose_w<<<cdiv(F0 * F1, B), B, 0, stream>>>(W1, W1t, F0, F1);
  k_transpose_w<<<cdiv(F1 * F2, B), B, 0, stream>>>(W2, W2t, F1, F2);

  // ================= layer 1 =================
  k_cvt_f16<<<cdiv(NN * F0, B), B, 0, stream>>>(x, ah, NN * F0);
  {
    int waves = (NN / 16) * (F1 / 64);
    k_wmma_gemm<128><<<cdiv(waves, 8), B, 0, stream>>>(ah, W1t, feat, NN, F1);
  }
  k_attn_coef<<<cdiv(NN * NH, B), B, 0, stream>>>(feat, al1, ar1, el, er, D1);
  k_init<<<cdiv(NN * F1, B), B, 0, stream>>>(hbuf, zb, mu, NN * F1);
  k_edge_max<<<cdiv(NE * NH, B), B, 0, stream>>>(src, dst, el, er, mu);
  k_edge_exp<<<cdiv(NE * NH, B), B, 0, stream>>>(src, dst, el, er, mu, ee, zb);
  k_edge_agg<<<NE, F1, 0, stream>>>(src, dst, feat, ee, zb, hbuf, D1);
  k_bias_elu_cvt<<<cdiv(NN * F1, B), B, 0, stream>>>(hbuf, b1, ah, NN * F1, F1);

  // ================= layer 2 =================
  {
    int waves = (NN / 16) * (F2 / 64);
    k_wmma_gemm<256><<<cdiv(waves, 8), B, 0, stream>>>(ah, W2t, feat, NN, F2);
  }
  k_attn_coef<<<cdiv(NN * NH, B), B, 0, stream>>>(feat, al2, ar2, el, er, D2);
  k_init<<<cdiv(NN * F2, B), B, 0, stream>>>(hbuf, zb, mu, NN * F2);   // hbuf now agg2
  k_edge_max<<<cdiv(NE * NH, B), B, 0, stream>>>(src, dst, el, er, mu);
  k_edge_exp<<<cdiv(NE * NH, B), B, 0, stream>>>(src, dst, el, er, mu, ee, zb);
  k_edge_agg<<<NE, F2, 0, stream>>>(src, dst, feat, ee, zb, hbuf, D2);

  // -------- head mean + bias --------
  k_head_mean<<<cdiv(NN * D2, B), B, 0, stream>>>(hbuf, b2, out, D2);
}
